// local_graph_32495722561768
// MI455X (gfx1250) — matrix operand order repaired and checked by reading, compiled, not run on hardware
//
#include <hip/hip_runtime.h>
#include <math.h>

// ---------------------------------------------------------------------------
// Sizes from the reference
// ---------------------------------------------------------------------------
#define NN     4096
#define KEDGE  64
#define EE     (NN * KEDGE)        // 262144 edges
#define HH     256
#define MTILE  128                 // rows per block (8 waves x 16 rows)
#define NBLK   (EE / MTILE)        // 2048 blocks
#define BN_EPS 1e-5f

typedef __attribute__((ext_vector_type(16))) _Float16 v16h;
typedef __attribute__((ext_vector_type(8)))  float    v8f;
typedef int b128_t __attribute__((vector_size(16)));   // payload type of async b128 ops

#define AS1 __attribute__((address_space(1)))
#define AS3 __attribute__((address_space(3)))

// gfx1250 async global->LDS path (ASYNCcnt), guarded so the file still
// compiles (and the host pass parses) when the builtins are not declared.
#if defined(__AMDGCN__) && \
    __has_builtin(__builtin_amdgcn_global_load_async_to_lds_b128) && \
    __has_builtin(__builtin_amdgcn_s_wait_asynccnt)
#define USE_ASYNC_LDS 1
#else
#define USE_ASYNC_LDS 0
#endif

// ---------------------------------------------------------------------------
// Workspace layout (bytes), total ~71.4 MB:
//   alpha     : NN*NN*4            = 67108864   @ 0
//   partials  : NBLK*512*4         = 4194304    @ 67108864
//   mu        : 256*4                           @ 71303168
//   rstd      : 256*4                           @ 71304192
//   W1T (f16) : 256*256*2          = 131072     @ 71305216
// lcker accumulates directly in d_out.
// ---------------------------------------------------------------------------
#define WS_ALPHA_OFF    0
#define WS_PART_OFF     67108864UL
#define WS_MU_OFF       71303168UL
#define WS_RSTD_OFF     71304192UL
#define WS_W1T_OFF      71305216UL

// ---------------------------------------------------------------------------
// Stage one 32-wide K-slab of W1T (f16, pre-transposed [n][k]) into one of
// two LDS buffers.  Async path: 4x global_load_async_to_lds_b128 per thread
// (64B), tracked by ASYNCcnt, no VGPR round-trip.
// ---------------------------------------------------------------------------
__device__ __forceinline__ void stage_slab(const _Float16* __restrict__ W1T,
                                           _Float16 (*Bt)[32],
                                           int buf, int ks, int tid)
{
#if USE_ASYNC_LDS
    AS1 b128_t* g = (AS1 b128_t*)(W1T + (size_t)tid * 256 + ks * 32);
    AS3 b128_t* l = (AS3 b128_t*)(&Bt[buf * 256 + tid][0]);
    __builtin_amdgcn_global_load_async_to_lds_b128(g,     l,     0, 0);
    __builtin_amdgcn_global_load_async_to_lds_b128(g + 1, l + 1, 0, 0);
    __builtin_amdgcn_global_load_async_to_lds_b128(g + 2, l + 2, 0, 0);
    __builtin_amdgcn_global_load_async_to_lds_b128(g + 3, l + 3, 0, 0);
#else
    const uint4* src = (const uint4*)(W1T + (size_t)tid * 256 + ks * 32);
    uint4* dst = (uint4*)(&Bt[buf * 256 + tid][0]);
    dst[0] = src[0]; dst[1] = src[1]; dst[2] = src[2]; dst[3] = src[3];
    if (ks < 7)
        __builtin_prefetch(W1T + (size_t)tid * 256 + (ks + 1) * 32, 0, 3);
#endif
}

// ---------------------------------------------------------------------------
// Shared WMMA core: each block computes a 128x256 tile of
//   h1 = (tanh(coor) @ W0 + b0) @ W1        (pre-b1, f32 accumulators)
// Wave w owns rows [16w,16w+16); acc[t] is the 16x16 C tile for columns
// [16t,16t+16).  K-loop: 8 steps of 32 using v_wmma_f32_16x16x32_f16, with
// double-buffered async LDS staging of the W1 slab and software-pipelined
// B-fragment loads.
// ---------------------------------------------------------------------------
__device__ __forceinline__ void gemm_tiles(
    const float* __restrict__ coor,
    const _Float16* __restrict__ W1T,     // [n][k] f16, 256x256
    _Float16 (*Bt)[32],                   // LDS: [2][256 cols][32 k] f16
    const float* W0s,                     // LDS: 4x256 (row-major d,k)
    const float* b0s,                     // LDS: 256
    v8f* acc)                             // [16] accumulators
{
    const int tid  = threadIdx.x;
    const int lane = tid & 31;
    const int hi   = lane >> 4;           // which 16-lane half
    const int rw   = lane & 15;           // row within wave tile / col within N tile
    const int wave = tid >> 5;

    const long e = (long)blockIdx.x * MTILE + wave * 16 + rw;
    const float t0 = tanhf(coor[e * 4 + 0]);
    const float t1 = tanhf(coor[e * 4 + 1]);
    const float t2 = tanhf(coor[e * 4 + 2]);
    const float t3 = tanhf(coor[e * 4 + 3]);

    #pragma unroll
    for (int t = 0; t < 16; ++t) { v8f z = {}; acc[t] = z; }

    // Prologue: start slab 0 into buffer 0.
    stage_slab(W1T, Bt, 0, 0, tid);

    for (int ks = 0; ks < 8; ++ks) {
        const int buf = ks & 1;
#if USE_ASYNC_LDS
        __builtin_amdgcn_s_wait_asynccnt(0);   // my async writes landed in LDS
#endif
        __syncthreads();                       // everyone's writes visible
        if (ks < 7)                            // overlap next slab with compute
            stage_slab(W1T, Bt, buf ^ 1, ks + 1, tid);

        // Build A fragment (contiguous K order, matching B):
        //  lane(lo): K = 0..15 ; lane(hi): K = 16..31
        v16h a;
        #pragma unroll
        for (int i = 0; i < 16; ++i) {
            const int kl = i + 16 * hi;
            const int kg = ks * 32 + kl;
            const float h0 = b0s[kg] + t0 * W0s[kg]       + t1 * W0s[256 + kg]
                                     + t2 * W0s[512 + kg] + t3 * W0s[768 + kg];
            a[i] = (_Float16)h0;
        }

        // 16 N-tiles; B fragment = 16 contiguous f16 (32B LDS load).
        // Software pipeline: load tile t+1 while WMMA t executes.
        const _Float16* base = &Bt[buf * 256][0] + rw * 32 + hi * 16;
        v16h bcur = *(const v16h*)(base);
        #pragma unroll
        for (int t = 0; t < 16; ++t) {
            v16h bnxt = bcur;
            if (t < 15) bnxt = *(const v16h*)(base + (t + 1) * 512);
            acc[t] = __builtin_amdgcn_wmma_f32_16x16x32_f16(
                false, a, false, bcur, (short)0, acc[t], false, false);
            bcur = bnxt;
        }
    }
}

// ---------------------------------------------------------------------------
// Pass 1: per-block column sums of t = tanh(h1 + b1) -> partials
// ---------------------------------------------------------------------------
__global__ void __launch_bounds__(256, 1)
stats_kernel(const float* __restrict__ coor, const float* __restrict__ W0,
             const float* __restrict__ b0, const _Float16* __restrict__ W1T,
             const float* __restrict__ b1, float* __restrict__ partials)
{
    __shared__ _Float16 Bt[2 * 256][32];
    __shared__ float W0s[1024];
    __shared__ float b0s[256];
    __shared__ float s1[256];
    __shared__ float s2[256];

    const int tid = threadIdx.x;
    for (int j = tid; j < 1024; j += 256) W0s[j] = W0[j];
    b0s[tid] = b0[tid];
    s1[tid] = 0.f; s2[tid] = 0.f;

    v8f acc[16];
    gemm_tiles(coor, W1T, Bt, W0s, b0s, acc);

    const int lane = tid & 31;
    const int rw   = lane & 15;
    #pragma unroll
    for (int t = 0; t < 16; ++t) {
        const int j = t * 16 + rw;
        const float b1j = b1[j];
        float s = 0.f, q = 0.f;
        #pragma unroll
        for (int r = 0; r < 8; ++r) {
            const float v = tanhf(acc[t][r] + b1j);
            s += v; q += v * v;
        }
        atomicAdd(&s1[j], s);   // ds_add_f32
        atomicAdd(&s2[j], q);
    }
    __syncthreads();
    partials[(size_t)blockIdx.x * 512 + tid]       = s1[tid];
    partials[(size_t)blockIdx.x * 512 + 256 + tid] = s2[tid];
}

__global__ void reduce_stats(const float* __restrict__ partials,
                             float* __restrict__ mu, float* __restrict__ rstd)
{
    const int j = threadIdx.x;   // 0..255
    float s = 0.f, q = 0.f;
    for (int b = 0; b < NBLK; ++b) {
        s += partials[(size_t)b * 512 + j];
        q += partials[(size_t)b * 512 + 256 + j];
    }
    const float m = s * (1.0f / (float)EE);
    const float v = q * (1.0f / (float)EE) - m * m;
    mu[j]   = m;
    rstd[j] = rsqrtf(v + BN_EPS);
}

// ---------------------------------------------------------------------------
// Pass 2: recompute h1, apply BN -> tanh -> @W2 -> relu, scatter into lcker
// ---------------------------------------------------------------------------
__global__ void __launch_bounds__(256, 1)
value_kernel(const float* __restrict__ coor, const float* __restrict__ W0,
             const float* __restrict__ b0, const _Float16* __restrict__ W1T,
             const float* __restrict__ b1, const float* __restrict__ gamma,
             const float* __restrict__ beta, const float* __restrict__ W2,
             const float* __restrict__ b2, const float* __restrict__ mu,
             const float* __restrict__ rstd, const int* __restrict__ sidx,
             float* __restrict__ lcker)
{
    __shared__ _Float16 Bt[2 * 256][32];
    __shared__ float W0s[1024];
    __shared__ float b0s[256];
    __shared__ float rowpart[128][17];   // +1 pad vs bank conflicts

    const int tid = threadIdx.x;
    for (int j = tid; j < 1024; j += 256) W0s[j] = W0[j];
    b0s[tid] = b0[tid];

    v8f acc[16];
    gemm_tiles(coor, W1T, Bt, W0s, b0s, acc);

    const int lane = tid & 31;
    const int hi   = lane >> 4;
    const int rw   = lane & 15;
    const int wave = tid >> 5;

    float racc[8];
    #pragma unroll
    for (int r = 0; r < 8; ++r) racc[r] = 0.f;

    #pragma unroll
    for (int t = 0; t < 16; ++t) {
        const int j  = t * 16 + rw;
        const float b1j = b1[j];
        const float gj  = gamma[j];
        const float bej = beta[j];
        const float mj  = mu[j];
        const float rj  = rstd[j];
        const float wj  = W2[j];
        #pragma unroll
        for (int r = 0; r < 8; ++r) {
            const float v  = tanhf(acc[t][r] + b1j);
            const float hn = gj * (v - mj) * rj + bej;
            racc[r] += tanhf(hn) * wj;
        }
    }
    // row M = r + 8*hi within the wave's 16-row tile; one writer per slot
    #pragma unroll
    for (int r = 0; r < 8; ++r)
        rowpart[wave * 16 + hi * 8 + r][rw] = racc[r];
    __syncthreads();

    if (tid < 128) {
        float s = 0.f;
        #pragma unroll
        for (int c = 0; c < 16; ++c) s += rowpart[tid][c];
        s += b2[0];
        s = fmaxf(s, 0.f);                       // relu
        const long e = (long)blockIdx.x * MTILE + tid;
        const int r = sidx[e];                   // row index (== e/64)
        const int c = sidx[(size_t)EE + e];      // col index
        atomicAdd(&lcker[(size_t)r * NN + c], s);
    }
}

// ---------------------------------------------------------------------------
// Attention: att[e] = | (center@Wq+bq) . (sphere@Wk+bk) |, scatter into alpha
// ---------------------------------------------------------------------------
__global__ void attn_kernel(const float* __restrict__ coor,
                            const float* __restrict__ Wq, const float* __restrict__ bq,
                            const float* __restrict__ Wk, const float* __restrict__ bk,
                            const int* __restrict__ sidx, float* __restrict__ alpha)
{
    const long e = (long)blockIdx.x * blockDim.x + threadIdx.x;
    if (e >= (long)EE) return;
    const long n  = e >> 6;            // node
    const long c0 = n << 6;            // center edge index = n*K
    const float s0 = coor[e * 4 + 0] + coor[e * 4 + 2];
    const float s1 = coor[e * 4 + 1] + coor[e * 4 + 3];
    const float q0 = coor[c0 * 4 + 0] + coor[c0 * 4 + 2];
    const float q1 = coor[c0 * 4 + 1] + coor[c0 * 4 + 3];
    float dot = 0.f;
    #pragma unroll
    for (int j = 0; j < 8; ++j) {
        const float qv = q0 * Wq[j] + q1 * Wq[8 + j] + bq[j];
        const float kv = s0 * Wk[j] + s1 * Wk[8 + j] + bk[j];
        dot += qv * kv;
    }
    const int r = sidx[e];
    const int c = sidx[(size_t)EE + e];
    atomicAdd(&alpha[(size_t)r * NN + c], fabsf(dot));
}

// ---------------------------------------------------------------------------
// Utility + finalize
// ---------------------------------------------------------------------------
__global__ void zero2_kernel(float* __restrict__ a, float* __restrict__ b, long n)
{
    const long i = (long)blockIdx.x * blockDim.x + threadIdx.x;
    if (i < n) { a[i] = 0.f; b[i] = 0.f; }
}

__global__ void prep_w1t_kernel(const float* __restrict__ W1, _Float16* __restrict__ W1T)
{
    const int idx = blockIdx.x * 256 + threadIdx.x;   // 0..65535
    const int n = idx >> 8, k = idx & 255;
    W1T[(size_t)n * 256 + k] = (_Float16)W1[(size_t)k * 256 + n];
}

__global__ void finalize_kernel(const float* __restrict__ alpha,
                                const float* __restrict__ geo,
                                const float* __restrict__ ang,
                                float* __restrict__ out, long n)
{
    const long i = (long)blockIdx.x * blockDim.x + threadIdx.x;
    if (i < n) {
        const float lck = out[i];
        out[i] = lck * expf(-alpha[i] * geo[i]) * ang[i];
    }
}

// ---------------------------------------------------------------------------
extern "C" void kernel_launch(void* const* d_in, const int* in_sizes, int n_in,
                              void* d_out, int out_size, void* d_ws, size_t ws_size,
                              hipStream_t stream)
{
    (void)in_sizes; (void)n_in; (void)out_size; (void)ws_size;

    const float* coor  = (const float*)d_in[1];
    const int*   sidx  = (const int*)d_in[2];
    const float* geo   = (const float*)d_in[3];
    const float* ang   = (const float*)d_in[4];
    const float* W0    = (const float*)d_in[5];
    const float* b0    = (const float*)d_in[6];
    const float* W1    = (const float*)d_in[7];
    const float* b1    = (const float*)d_in[8];
    const float* gamma = (const float*)d_in[9];
    const float* beta  = (const float*)d_in[10];
    const float* W2    = (const float*)d_in[11];
    const float* b2    = (const float*)d_in[12];
    const float* Wq    = (const float*)d_in[13];
    const float* bq    = (const float*)d_in[14];
    const float* Wk    = (const float*)d_in[15];
    const float* bk    = (const float*)d_in[16];

    float* out = (float*)d_out;
    char*  ws  = (char*)d_ws;
    float*    alpha    = (float*)(ws + WS_ALPHA_OFF);
    float*    partials = (float*)(ws + WS_PART_OFF);
    float*    mu       = (float*)(ws + WS_MU_OFF);
    float*    rstd     = (float*)(ws + WS_RSTD_OFF);
    _Float16* W1T      = (_Float16*)(ws + WS_W1T_OFF);

    const long nn = (long)NN * NN;

    zero2_kernel<<<(int)((nn + 255) / 256), 256, 0, stream>>>(alpha, out, nn);
    prep_w1t_kernel<<<256, 256, 0, stream>>>(W1, W1T);
    stats_kernel<<<NBLK, 256, 0, stream>>>(coor, W0, b0, W1T, b1, partials);
    reduce_stats<<<1, 256, 0, stream>>>(partials, mu, rstd);
    value_kernel<<<NBLK, 256, 0, stream>>>(coor, W0, b0, W1T, b1, gamma, beta,
                                           W2, b2, mu, rstd, sidx, out);
    attn_kernel<<<EE / 256, 256, 0, stream>>>(coor, Wq, bq, Wk, bk, sidx, alpha);
    finalize_kernel<<<(int)((nn + 255) / 256), 256, 0, stream>>>(alpha, geo, ang, out, nn);
}
